// MultiGraphConvolution_Layer_5712306504000
// MI455X (gfx1250) — compile-verified, matrix operand order checked by reading, not run on hardware
//
#include <hip/hip_runtime.h>

// MultiGraphConvolution fused layer for MI455X (gfx1250, wave32, WMMA).
// out[n,o] = sum_s attn[s]*(X[s]@W[s])[n,o] + bias[o]
// == single GEMM M=200000, K=1024, N=128 with attn folded into W.
// bf16 hi/lo split (3 WMMA terms) for ~fp32 accuracy at bf16 matrix rate.

typedef __attribute__((ext_vector_type(16))) __bf16 v16bf;
typedef __attribute__((ext_vector_type(8)))  __bf16 v8bf;
typedef __attribute__((ext_vector_type(8)))  float  v8f;

#define SDIM   8
#define NROWS  200000
#define KF     128       // inner dim per s
#define OF     128       // output features
#define LROW   136       // padded LDS row stride (elements); 136*2=272B, 16B aligned

// Per-lane WMMA 16-bit operand fetch from LDS:
// lane holds elements [e0 .. e0+7] (VGPRs 0-3) and [e0+16 .. e0+23] (VGPRs 4-7)
// of its row, where e0 = chunk_base + 8*(lane>=16).  Two ds_load_b128.
__device__ __forceinline__ v16bf lds_ld16(const __bf16* base, int e0) {
  v8bf x = *(const v8bf*)(base + e0);
  v8bf y = *(const v8bf*)(base + e0 + 16);
  return __builtin_shufflevector(x, y, 0,1,2,3,4,5,6,7,8,9,10,11,12,13,14,15);
}

__device__ __forceinline__ void split_bf16(float v, __bf16& h, __bf16& l) {
  h = (__bf16)v;
  l = (__bf16)(v - (float)h);
}

__global__ __launch_bounds__(256)
void mgc_wmma_bf16x3_kernel(const float* __restrict__ X,
                            const float* __restrict__ W,
                            const float* __restrict__ attn,
                            const float* __restrict__ bias,
                            float* __restrict__ out) {
  extern __shared__ __attribute__((aligned(16))) __bf16 smem[];
  __bf16* bhi = smem;                      // [OF][LROW]  (col-major: [n][k])
  __bf16* blo = bhi + OF * LROW;
  __bf16* ahi = blo + OF * LROW;           // 8 waves * 32 rows * LROW
  __bf16* alo = ahi + 8 * 32 * LROW;

  const int tid  = threadIdx.x;
  const int wv   = tid >> 5;
  const int lane = tid & 31;
  const int m    = lane & 15;              // M (A/C) or N (B) index in tile
  const int koff = (lane >> 4) << 3;       // 0 or 8: per-lane K base

  const long long rowbase = (long long)blockIdx.x * 256 + (long long)wv * 32;

  v8f acc[2][8];
#pragma unroll
  for (int t = 0; t < 2; ++t)
#pragma unroll
    for (int nt = 0; nt < 8; ++nt) acc[t][nt] = (v8f)0.0f;

  __bf16* aw_hi = ahi + wv * 32 * LROW;
  __bf16* aw_lo = alo + wv * 32 * LROW;

  for (int s = 0; s < SDIM; ++s) {
    __syncthreads();   // previous phase's LDS consumers done

    // ---- stage B: W'[s] = attn[s]*W[s], split hi/lo, transpose to [n][k] ----
    {
      const float asc = attn[s];
      const float* wp = W + s * KF * OF;
      for (int i = tid; i < KF * OF; i += 256) {
        const int k = i >> 7;        // W[s] is [k][o] row-major
        const int o = i & (OF - 1);
        __bf16 h, l;
        split_bf16(wp[i] * asc, h, l);
        bhi[o * LROW + k] = h;
        blo[o * LROW + k] = l;
      }
    }

    // ---- stage A: this wave's 32 rows of X[s], split hi/lo into LDS ----
    {
      const float* xs = X + (long long)s * NROWS * KF;
      for (int r = 0; r < 32; ++r) {
        long long row = rowbase + r;
        if (row >= NROWS) row = NROWS - 1;   // clamp; stores are guarded
        const float* xp = xs + row * KF;
#pragma unroll
        for (int j = 0; j < 4; ++j) {
          const int k = lane + j * 32;       // coalesced 128B reads
          __bf16 h, l;
          split_bf16(xp[k], h, l);
          aw_hi[r * LROW + k] = h;
          aw_lo[r * LROW + k] = l;
        }
      }
    }
    __syncthreads();

    // ---- compute: 4 K-chunks of 32, 8 N-tiles, 2 M-tiles, 3 precision terms
    for (int kc = 0; kc < 4; ++kc) {
      const int ke = kc * 32 + koff;
      v16bf a0h = lds_ld16(aw_hi, (0 * 16 + m) * LROW + ke);
      v16bf a0l = lds_ld16(aw_lo, (0 * 16 + m) * LROW + ke);
      v16bf a1h = lds_ld16(aw_hi, (1 * 16 + m) * LROW + ke);
      v16bf a1l = lds_ld16(aw_lo, (1 * 16 + m) * LROW + ke);
#pragma unroll
      for (int nt = 0; nt < 8; ++nt) {
        const int n = nt * 16 + m;
        v16bf bh = lds_ld16(bhi, n * LROW + ke);
        v16bf bl = lds_ld16(blo, n * LROW + ke);
        acc[0][nt] = __builtin_amdgcn_wmma_f32_16x16x32_bf16(
            false, a0h, false, bh, (short)0, acc[0][nt], false, false);
        acc[0][nt] = __builtin_amdgcn_wmma_f32_16x16x32_bf16(
            false, a0h, false, bl, (short)0, acc[0][nt], false, false);
        acc[0][nt] = __builtin_amdgcn_wmma_f32_16x16x32_bf16(
            false, a0l, false, bh, (short)0, acc[0][nt], false, false);
        acc[1][nt] = __builtin_amdgcn_wmma_f32_16x16x32_bf16(
            false, a1h, false, bh, (short)0, acc[1][nt], false, false);
        acc[1][nt] = __builtin_amdgcn_wmma_f32_16x16x32_bf16(
            false, a1h, false, bl, (short)0, acc[1][nt], false, false);
        acc[1][nt] = __builtin_amdgcn_wmma_f32_16x16x32_bf16(
            false, a1l, false, bh, (short)0, acc[1][nt], false, false);
      }
    }
  }

  // ---- epilogue: += bias, store. C layout: elem r -> M = r + 8*(lane>=16) ----
  float bv[8];
#pragma unroll
  for (int nt = 0; nt < 8; ++nt) bv[nt] = bias[nt * 16 + m];

  const int rhalf = (lane >> 4) * 8;
#pragma unroll
  for (int t = 0; t < 2; ++t) {
#pragma unroll
    for (int rr = 0; rr < 8; ++rr) {
      const long long row = rowbase + t * 16 + rhalf + rr;
      if (row < NROWS) {
        float* op = out + row * OF;
#pragma unroll
        for (int nt = 0; nt < 8; ++nt)
          op[nt * 16 + m] = acc[t][nt][rr] + bv[nt];
      }
    }
  }
}

extern "C" void kernel_launch(void* const* d_in, const int* in_sizes, int n_in,
                              void* d_out, int out_size, void* d_ws, size_t ws_size,
                              hipStream_t stream) {
  (void)in_sizes; (void)n_in; (void)d_ws; (void)ws_size; (void)out_size;
  const float* X    = (const float*)d_in[0];   // (8, 200000, 128) f32
  const float* W    = (const float*)d_in[1];   // (8, 128, 128)    f32
  const float* attn = (const float*)d_in[2];   // (8,)             f32
  const float* bias = (const float*)d_in[3];   // (128,)           f32
  float* out = (float*)d_out;                  // (200000, 128)    f32

  const int rows_per_wg = 256;                 // 8 waves * 32 rows
  const int grid = (NROWS + rows_per_wg - 1) / rows_per_wg;   // 782
  // LDS: B hi/lo (2*128*136) + A hi/lo (2*8*32*136) bf16 = 208896 bytes
  const size_t smem = (size_t)(2 * OF * LROW + 2 * 8 * 32 * LROW) * sizeof(__bf16);

  mgc_wmma_bf16x3_kernel<<<grid, 256, smem, stream>>>(X, W, attn, bias, out);
}
